// Top_K_MoEAdapter_33724083208859
// MI455X (gfx1250) — matrix operand order repaired and checked by reading, not compiled, not run on hardware
//
#include <hip/hip_runtime.h>

// ---------------- problem constants ----------------
constexpr int D_IN = 2048;   // IN_DIM
constexpr int BOT  = 512;    // BOT_DIM
constexpr int NEXP = 9;      // 8 routed experts + 1 shared adapter
constexpr int KD   = 2080;   // down K: 2048 + 1 bias col, padded to mult of 32
constexpr int KSD  = KD / 32;    // 65 k-steps (down GEMM)
constexpr int KU   = 4640;   // up K: 9*512 + 16 bias cols, padded to mult of 32
constexpr int KSU  = KU / 32;    // 145 k-steps (up GEMM)
constexpr int ND   = NEXP * BOT; // 4608 down-GEMM N
constexpr int NTU  = D_IN / 16;  // 128 up n-tiles

// ---------------- WMMA types ----------------
typedef __attribute__((ext_vector_type(16))) __bf16    v16bf;
typedef __attribute__((ext_vector_type(8)))  float     v8f;
typedef __attribute__((ext_vector_type(4)))  unsigned  u32x4;

union Frag { u32x4 u[2]; v16bf v; };

// async copy global -> LDS (16B per lane), tracked by ASYNCcnt
__device__ __forceinline__ void async_ld16(void* lds_ptr, const __bf16* g) {
  unsigned loff = (unsigned)(unsigned long long)lds_ptr;  // low 32 bits = LDS offset
  asm volatile("global_load_async_to_lds_b128 %0, %1, off"
               :: "v"(loff), "v"(g) : "memory");
}
__device__ __forceinline__ void wait_async0() {
  asm volatile("s_wait_asynccnt 0" ::: "memory");
}

// ---------------- 1) routing: softmax + top-2 + renorm ----------------
__global__ __launch_bounds__(256) void k_gate(const float* __restrict__ x,
                                              const float* __restrict__ gw,
                                              float* __restrict__ comb, int T) {
  int gtid = blockIdx.x * blockDim.x + threadIdx.x;
  int tok  = gtid >> 5;
  int lane = gtid & 31;
  if (tok >= T) return;
  const float* xr = x + (size_t)tok * D_IN;
  float s[8];
#pragma unroll
  for (int e = 0; e < 8; ++e) s[e] = 0.f;
  for (int k = lane; k < D_IN; k += 32) {
    float xv = xr[k];
#pragma unroll
    for (int e = 0; e < 8; ++e) s[e] = fmaf(xv, gw[e * D_IN + k], s[e]);
  }
#pragma unroll
  for (int off = 16; off > 0; off >>= 1) {
#pragma unroll
    for (int e = 0; e < 8; ++e) s[e] += __shfl_xor(s[e], off, 32);
  }
  if (lane == 0) {
    float m = s[0];
#pragma unroll
    for (int e = 1; e < 8; ++e) m = fmaxf(m, s[e]);
    float p[8];
#pragma unroll
    for (int e = 0; e < 8; ++e) p[e] = __expf(s[e] - m);
    int i0 = 0;
#pragma unroll
    for (int e = 1; e < 8; ++e) if (p[e] > p[i0]) i0 = e;
    int i1 = (i0 == 0) ? 1 : 0;
#pragma unroll
    for (int e = 0; e < 8; ++e) if (e != i0 && p[e] > p[i1]) i1 = e;
    float r = p[i0] + p[i1];
    float* c = comb + (size_t)tok * 16;
#pragma unroll
    for (int e = 0; e < 8; ++e)
      c[e] = (e == i0) ? p[i0] / r : ((e == i1) ? p[i1] / r : 0.f);
    c[8] = 1.f;  // shared expert always on
#pragma unroll
    for (int e = 9; e < 16; ++e) c[e] = 0.f;
  }
}

// ---------------- 2) x -> bf16 with bias-1 column and zero pad ----------------
__global__ __launch_bounds__(256) void k_cvt_x(const float* __restrict__ x,
                                               __bf16* __restrict__ xb, int T) {
  size_t i = (size_t)blockIdx.x * 256 + threadIdx.x;
  size_t total = (size_t)T * KD;
  if (i >= total) return;
  size_t t = i / KD;
  int k = (int)(i % KD);
  float v = (k < D_IN) ? x[t * (size_t)D_IN + k] : (k == D_IN ? 1.f : 0.f);
  xb[i] = (__bf16)v;
}

// ---------------- 3) pack down weights (wd+bd, ws_d+bs_d) into B-fragments ----
// frag(ntile_global, kk): 32 lanes x 16 bf16 lane-major; lane<16: col=lane, K=s;
// lane>=16: col=lane-16, K=16+s (contiguous k order, matching the A reads); ntile_global = e*32+nt
__global__ __launch_bounds__(256) void k_pack_wd(const float* __restrict__ wd,
                                                 const float* __restrict__ bd,
                                                 const float* __restrict__ wsd,
                                                 const float* __restrict__ bsd,
                                                 __bf16* __restrict__ dst) {
  size_t total = (size_t)NEXP * 32 * KSD * 512;
  size_t i = (size_t)blockIdx.x * 256 + threadIdx.x;
  if (i >= total) return;
  int s    = (int)(i & 15);
  int lane = (int)((i >> 4) & 31);
  size_t frag = i >> 9;
  int kk = (int)(frag % KSD);
  size_t t2 = frag / KSD;
  int ntile = (int)(t2 % 32);
  int e = (int)(t2 / 32);
  int n = ntile * 16 + (lane & 15);
  int k = kk * 32 + (lane >> 4) * 16 + s;
  float v;
  if (k < D_IN)
    v = (e < 8) ? wd[((size_t)e * BOT + n) * D_IN + k] : wsd[(size_t)n * D_IN + k];
  else if (k == D_IN)
    v = (e < 8) ? bd[e * BOT + n] : bsd[n];
  else
    v = 0.f;
  dst[i] = (__bf16)v;
}

// ---------------- 4) pack up weights (wu+bu, ws_u+bs_u) into B-fragments ------
__global__ __launch_bounds__(256) void k_pack_wu(const float* __restrict__ wu,
                                                 const float* __restrict__ bu,
                                                 const float* __restrict__ wsu,
                                                 const float* __restrict__ bsu,
                                                 __bf16* __restrict__ dst) {
  size_t total = (size_t)NTU * KSU * 512;
  size_t i = (size_t)blockIdx.x * 256 + threadIdx.x;
  if (i >= total) return;
  int s    = (int)(i & 15);
  int lane = (int)((i >> 4) & 31);
  size_t frag = i >> 9;
  int kk = (int)(frag % KSU);
  int ntile = (int)(frag / KSU);
  int n = ntile * 16 + (lane & 15);     // output dim d
  int k = kk * 32 + (lane >> 4) * 16 + s;
  float v = 0.f;
  if (k < NEXP * BOT) {
    int e = k / BOT, b = k % BOT;
    v = (e < 8) ? wu[((size_t)e * D_IN + n) * BOT + b] : wsu[(size_t)n * BOT + b];
  } else if (k < NEXP * BOT + 16) {
    int eb = k - NEXP * BOT;
    if (eb < 8)       v = bu[(size_t)eb * D_IN + n];
    else if (eb == 8) v = bsu[n];
  }
  dst[i] = (__bf16)v;
}

// ---------------- 5) H bias columns: H[t, 4608+c] = comb[t, c] ----------------
__global__ __launch_bounds__(256) void k_hbias(const float* __restrict__ comb,
                                               __bf16* __restrict__ H, int T) {
  int i = blockIdx.x * 256 + threadIdx.x;
  if (i >= T * 32) return;
  int t = i >> 5, cc = i & 31;
  H[(size_t)t * KU + NEXP * BOT + cc] =
      (__bf16)(cc < 16 ? comb[(size_t)t * 16 + cc] : 0.f);
}

// ---------------- 6) 256x256-tile GEMM, async-LDS double buffered -------------
// MODE 0: C = gelu(A@B^T) * combine  -> bf16 H   (down)
// MODE 1: C = A@B^T                  -> fp32 out (up)
template <int MODE>
__global__ __launch_bounds__(512, 1) void k_gemm(
    const __bf16* __restrict__ A, int lda, int ksteps,
    const __bf16* __restrict__ Bz,
    const float* __restrict__ comb,
    void* __restrict__ Cv, int ldc) {
  __shared__ __align__(16) char smem[2 * 32768];  // [buf][A 16KB | B 16KB]
  const int tid  = threadIdx.x;
  const int wv   = tid >> 5, lane = tid & 31;
  const int l15  = lane & 15, half = lane >> 4;
  const int r    = wv >> 2, c = wv & 3;           // 4x4 wave grid
  const int tok0 = blockIdx.x * 256;
  const int n0   = blockIdx.y * 256;

  const __bf16* aglob = A + (size_t)tok0 * lda;

  auto stage = [&](int buf, int kk) {
#pragma unroll
    for (int it = 0; it < 2; ++it) {
      int idx = tid + it * 512;  // 1024 16B-chunks each for A and B
      {  // A chunk: rows 0..255 x 32 K, row-major (32 elems = 4 chunks/row)
        int row = idx >> 2, q = idx & 3;
        const __bf16* g = aglob + (size_t)row * lda + kk * 32 + q * 8;
        async_ld16(smem + buf * 32768 + (row * 32 + q * 8) * 2, g);
      }
      {  // B fragments: 16 n-tiles x 1KB (64 chunks each)
        int nt = idx >> 6, q = idx & 63;
        const __bf16* g = Bz + (((size_t)(n0 / 16 + nt)) * ksteps + kk) * 512 + q * 8;
        async_ld16(smem + buf * 32768 + 16384 + (nt * 512 + q * 8) * 2, g);
      }
    }
  };

  v8f acc[4][4] = {};
  stage(0, 0);
  wait_async0();
  __syncthreads();

  for (int kk = 0; kk < ksteps; ++kk) {
    int cur = kk & 1;
    if (kk + 1 < ksteps) stage(cur ^ 1, kk + 1);

    const unsigned* ad = (const unsigned*)(smem + cur * 32768);          // 16 dw/row
    const unsigned* bd = (const unsigned*)(smem + cur * 32768 + 16384);
    Frag a[4];
#pragma unroll
    for (int mt = 0; mt < 4; ++mt) {
      const u32x4* p = (const u32x4*)(ad + (r * 64 + mt * 16 + l15) * 16 + half * 8);
      a[mt].u[0] = p[0];  // K 0-7 (lanes<16) / 16-23 (lanes>=16)
      a[mt].u[1] = p[1];  // K 8-15 / 24-31
    }
#pragma unroll
    for (int nt = 0; nt < 4; ++nt) {
      Frag b;
      const u32x4* p = (const u32x4*)(bd + (c * 4 + nt) * 256 + lane * 8);
      b.u[0] = p[0];
      b.u[1] = p[1];
#pragma unroll
      for (int mt = 0; mt < 4; ++mt)
        acc[mt][nt] = __builtin_amdgcn_wmma_f32_16x16x32_bf16(
            false, a[mt].v, false, b.v, (short)0, acc[mt][nt], false, false);
    }
    wait_async0();
    __syncthreads();
  }

#pragma unroll
  for (int mt = 0; mt < 4; ++mt) {
#pragma unroll
    for (int nt = 0; nt < 4; ++nt) {
#pragma unroll
      for (int j = 0; j < 8; ++j) {
        int trow = tok0 + r * 64 + mt * 16 + j + 8 * half;  // C/D: M = j + 8*half
        int ncol = n0 + c * 64 + nt * 16 + l15;             // C/D: N = lane&15
        float v = acc[mt][nt][j];
        if constexpr (MODE == 0) {
          float cmb = comb[(size_t)trow * 16 + (ncol >> 9)];  // expert = n/512
          float g = 0.5f * v * (1.f + erff(v * 0.70710678118f));
          ((__bf16*)Cv)[(size_t)trow * ldc + ncol] = (__bf16)(g * cmb);
        } else {
          ((float*)Cv)[(size_t)trow * ldc + ncol] = v;
        }
      }
    }
  }
}

// ---------------- launcher ----------------
extern "C" void kernel_launch(void* const* d_in, const int* in_sizes, int n_in,
                              void* d_out, int out_size, void* d_ws, size_t ws_size,
                              hipStream_t stream) {
  const float* x   = (const float*)d_in[0];
  const float* gw  = (const float*)d_in[1];
  const float* wd  = (const float*)d_in[2];
  const float* bd  = (const float*)d_in[3];
  const float* wu  = (const float*)d_in[4];
  const float* bu  = (const float*)d_in[5];
  const float* wsd = (const float*)d_in[6];
  const float* bsd = (const float*)d_in[7];
  const float* wsu = (const float*)d_in[8];
  const float* bsu = (const float*)d_in[9];
  float* out = (float*)d_out;
  const int T = in_sizes[0] / D_IN;  // 8192

  char* ws = (char*)d_ws;
  size_t off = 0;
  float*  comb = (float*)(ws + off);  off += (size_t)T * 16 * sizeof(float);
  __bf16* xbp  = (__bf16*)(ws + off); off += (size_t)T * KD * 2;
  __bf16* wdz  = (__bf16*)(ws + off); off += (size_t)NEXP * 32 * KSD * 512 * 2;
  __bf16* wuz  = (__bf16*)(ws + off); off += (size_t)NTU * KSU * 512 * 2;
  __bf16* H    = (__bf16*)(ws + off); off += (size_t)T * KU * 2;
  (void)ws_size;

  k_gate<<<(T * 32 + 255) / 256, 256, 0, stream>>>(x, gw, comb, T);

  size_t nx = (size_t)T * KD;
  k_cvt_x<<<(int)((nx + 255) / 256), 256, 0, stream>>>(x, xbp, T);

  size_t nwd = (size_t)NEXP * 32 * KSD * 512;
  k_pack_wd<<<(int)((nwd + 255) / 256), 256, 0, stream>>>(wd, bd, wsd, bsd, wdz);

  size_t nwu = (size_t)NTU * KSU * 512;
  k_pack_wu<<<(int)((nwu + 255) / 256), 256, 0, stream>>>(wu, bu, wsu, bsu, wuz);

  k_hbias<<<(T * 32 + 255) / 256, 256, 0, stream>>>(comb, H, T);

  // down: [T,2080] @ [4608,2080]^T -> H[T,4640] (gelu * combine epilogue)
  k_gemm<0><<<dim3(T / 256, ND / 256), 512, 0, stream>>>(
      xbp, KD, KSD, wdz, comb, (void*)H, KU);

  // up: [T,4640] @ [2048,4640]^T -> out[T,2048] (biases folded into K)
  k_gemm<1><<<dim3(T / 256, D_IN / 256), 512, 0, stream>>>(
      H, KU, KSU, wuz, nullptr, (void*)out, D_IN);
}
